// DGCNN_encoder_46042049413629
// MI455X (gfx1250) — matrix operand order repaired and checked by reading, not compiled, not run on hardware
//
#include <hip/hip_runtime.h>

typedef __attribute__((ext_vector_type(16))) __bf16 v16bf;
typedef __attribute__((ext_vector_type(8)))  __bf16 v8bf;
typedef __attribute__((ext_vector_type(8)))  float  v8f;

#define NPTS 1024
#define NB   8
#define KNN  20
#define SLOPE 0.2f

#if __has_builtin(__builtin_amdgcn_global_load_async_to_lds_b128) && \
    __has_builtin(__builtin_amdgcn_s_wait_asynccnt)
#define HAVE_ASYNC_LDS 1
typedef int b128_t __attribute__((vector_size(16)));
typedef __attribute__((address_space(1))) b128_t* g128p;   // global 16B vector ptr
typedef __attribute__((address_space(3))) b128_t* l128p;   // LDS    16B vector ptr
#else
#define HAVE_ASYNC_LDS 0
#endif

// ---------- helpers ----------
__device__ __forceinline__ unsigned short f2bf(float f) {
  union { float f; unsigned u; } v; v.f = f;
  unsigned u = v.u;
  u += 0x7FFFu + ((u >> 16) & 1u);          // round-to-nearest-even
  return (unsigned short)(u >> 16);
}
__device__ __forceinline__ unsigned fkey(float f) {   // monotone float->uint
  union { float f; unsigned u; } v; v.f = f;
  unsigned u = v.u;
  return u ^ ((u >> 31) ? 0xFFFFFFFFu : 0x80000000u);
}
__device__ __forceinline__ float unfkey(unsigned u) {
  u ^= ((u >> 31) ? 0x80000000u : 0xFFFFFFFFu);
  union { unsigned u; float f; } v; v.u = u;
  return v.f;
}

// A fragment: 16x32 bf16, row-major source W[o][K], per ISA 7.12.2 layout
__device__ __forceinline__ v16bf load_a_frag(const unsigned short* __restrict__ W,
                                             int ldk, int o, int kk, int lane) {
  const int hi = (lane >= 16) ? 8 : 0;
  const v8bf a0 = *(const v8bf*)(W + o * ldk + kk + hi);
  const v8bf a1 = *(const v8bf*)(W + o * ldk + kk + 16 + hi);
  v16bf a;
#pragma unroll
  for (int i = 0; i < 8; i++) { a[i] = a0[i]; a[8 + i] = a1[i]; }
  return a;
}
// B fragment: 32x16 bf16; base points at column's K-major row start + kk
__device__ __forceinline__ v16bf load_b_frag(const unsigned short* __restrict__ base,
                                             int lane) {
  const int hi = (lane >= 16) ? 8 : 0;
  const v8bf b0 = *(const v8bf*)(base + hi);
  const v8bf b1 = *(const v8bf*)(base + hi + 16);
  v16bf b;
#pragma unroll
  for (int i = 0; i < 8; i++) { b[i] = b0[i]; b[8 + i] = b1[i]; }
  return b;
}

// ---------- weight convert (f32 -> bf16, K padded to KP) ----------
__global__ void convw_kernel(const float* __restrict__ src, unsigned short* __restrict__ dst,
                             int O, int Cin, int KP) {
  int i = blockIdx.x * 256 + threadIdx.x;
  if (i < O * KP) {
    int o = i / KP, c = i % KP;
    float v = (c < Cin) ? src[o * Cin + c] : 0.f;
    dst[i] = f2bf(v);
  }
}

// ---------- kNN: async-LDS double-buffered candidate tiles, register top-20 ----------
template <int C>
__global__ __launch_bounds__(128) void knn_kernel(const float* __restrict__ xin, int ldb,
                                                  int* __restrict__ idx) {
  constexpr int TM = 32;
  constexpr int CH_TOT = C * (TM / 4);                  // b128 chunks per tile
  constexpr int CHUNKS = (CH_TOT + 127) / 128;          // async issues per thread-loop
  __shared__ float T[2][C][TM];
  const int b = blockIdx.y;
  const int n = blockIdx.x * 128 + threadIdx.x;
  const float* xb = xin + b * ldb;
  float q[C];
#pragma unroll
  for (int c = 0; c < C; c++) q[c] = xb[c * NPTS + n];
  float bv[KNN]; int bi[KNN];
#pragma unroll
  for (int k = 0; k < KNN; k++) { bv[k] = -3.4e38f; bi[k] = 0; }

#if HAVE_ASYNC_LDS
  auto issue_tile = [&](int buf, int m0) {
    for (int ch = threadIdx.x; ch < CH_TOT; ch += 128) {
      int c = ch / (TM / 4), qq = ch % (TM / 4);
      __builtin_amdgcn_global_load_async_to_lds_b128(
          (g128p)(xb + c * NPTS + m0 + qq * 4),
          (l128p)(&T[buf][c][qq * 4]), 0, 0);
    }
  };
  issue_tile(0, 0);
#endif

  for (int m0 = 0; m0 < NPTS; m0 += TM) {
    const int buf = (m0 / TM) & 1;
#if HAVE_ASYNC_LDS
    if (m0 + TM < NPTS) {
      issue_tile(buf ^ 1, m0 + TM);                     // prefetch next tile
      __builtin_amdgcn_s_wait_asynccnt(CHUNKS);         // retire current tile only
    } else {
      __builtin_amdgcn_s_wait_asynccnt(0);
    }
#else
    for (int t = threadIdx.x; t < C * TM; t += 128) {
      int c = t / TM, mm = t % TM;
      T[buf][c][mm] = xb[c * NPTS + m0 + mm];
    }
#endif
    __syncthreads();
    for (int mm = 0; mm < TM; mm++) {
      float d = 0.f;
#pragma unroll
      for (int c = 0; c < C; c++) { float t2 = q[c] - T[buf][c][mm]; d += t2 * t2; }
      float pd = -d;
      if (pd > bv[KNN - 1]) {            // branchless sorted insert (carry-down)
        float cv = pd; int ci = m0 + mm;
#pragma unroll
        for (int j = 0; j < KNN; j++) {
          bool take = cv > bv[j];
          float tv = take ? cv : bv[j];  int ti = take ? ci : bi[j];
          float nv = take ? bv[j] : cv;  int ni = take ? bi[j] : ci;
          bv[j] = tv; bi[j] = ti; cv = nv; ci = ni;
        }
      }
    }
    __syncthreads();                     // tile consumed before its buffer is refilled
  }
#pragma unroll
  for (int k = 0; k < KNN; k++) idx[(b * NPTS + n) * KNN + k] = bi[k];
}

// ---------- EdgeConv GEMM (mode 0: BN stats, mode 1: BN+LReLU+max-over-k) ----------
template <int C, int KP, int O>
__global__ __launch_bounds__(256) void edge_gemm_kernel(
    const float* __restrict__ xin, int ldb, const int* __restrict__ idx,
    const unsigned short* __restrict__ Wbf,
    const float* __restrict__ scale, const float* __restrict__ shift,
    float* __restrict__ xout, int ldbo, int chOff,
    float* __restrict__ gsum, float* __restrict__ gssq, int mode) {
  constexpr int NPB = 4, COLS = NPB * KNN;          // 80 columns, 5 col tiles
  __shared__ unsigned short BfT[COLS][KP + 8];      // K-major per column
  __shared__ float lsum[O], lssq[O];
  __shared__ unsigned lmax[O * NPB];

  const int b = blockIdx.y, n0 = blockIdx.x * NPB, tid = threadIdx.x;
  if (mode == 0) { for (int t = tid; t < O; t += 256) { lsum[t] = 0.f; lssq[t] = 0.f; } }
  else           { for (int t = tid; t < O * NPB; t += 256) lmax[t] = 0u; }

  const float* xb = xin + b * ldb;
  for (int e = tid; e < KP * COLS; e += 256) {      // gather graph feature -> bf16 LDS
    int c = e / COLS, col = e % COLS;
    int p = col / KNN, k = col % KNN, n = n0 + p;
    float v = 0.f;
    if (c < C) {
      int m = idx[(b * NPTS + n) * KNN + k];
      v = xb[c * NPTS + m] - xb[c * NPTS + n];
    } else if (c < 2 * C) {
      v = xb[(c - C) * NPTS + n];
    }
    BfT[col][c] = f2bf(v);
  }
  __syncthreads();

  const int lane = tid & 31, wave = tid >> 5, r = lane & 15;
  const int hi8 = (lane >= 16) ? 8 : 0;
  constexpr int RT = O / 16, CT = COLS / 16, TT = RT * CT;
  for (int t = wave; t < TT; t += 8) {
    int rt = t / CT, ct = t % CT, col = ct * 16 + r;
    v8f acc = {};
#pragma unroll
    for (int kk = 0; kk < KP; kk += 32) {
      v16bf a = load_a_frag(Wbf, KP, rt * 16 + r, kk, lane);
      v16bf bb = load_b_frag(&BfT[col][kk], lane);
      acc = __builtin_amdgcn_wmma_f32_16x16x32_bf16(false, a, false, bb, (short)0, acc,
                                                    false, false);
    }
    if (mode == 0) {
#pragma unroll
      for (int j = 0; j < 8; j++) {
        int o = rt * 16 + j + hi8;
        float v = acc[j];
        atomicAdd(&lsum[o], v);
        atomicAdd(&lssq[o], v * v);
      }
    } else {
      int p = col / KNN;
#pragma unroll
      for (int j = 0; j < 8; j++) {
        int o = rt * 16 + j + hi8;
        float v = acc[j] * scale[o] + shift[o];
        v = v > 0.f ? v : SLOPE * v;
        atomicMax(&lmax[o * NPB + p], fkey(v));
      }
    }
  }
  __syncthreads();
  if (mode == 0) {
    for (int t = tid; t < O; t += 256) {
      atomicAdd(&gsum[t], lsum[t]);
      atomicAdd(&gssq[t], lssq[t]);
    }
  } else {
    for (int t = tid; t < O * NPB; t += 256) {
      int o = t / NPB, p = t % NPB;
      xout[b * ldbo + (chOff + o) * NPTS + n0 + p] = unfkey(lmax[t]);
    }
  }
}

// ---------- BN finalize: scale = g*rsqrt(var+eps), shift = b - mu*scale ----------
__global__ void finalize_kernel(const float* __restrict__ gsum, const float* __restrict__ gssq,
                                const float* __restrict__ g, const float* __restrict__ bet,
                                float* __restrict__ scale, float* __restrict__ shift,
                                int O, float invcnt) {
  int o = blockIdx.x * 256 + threadIdx.x;
  if (o < O) {
    float mu = gsum[o] * invcnt;
    float var = gssq[o] * invcnt - mu * mu;
    float rs = rsqrtf(var + 1e-5f);
    float sc = g[o] * rs;
    scale[o] = sc;
    shift[o] = bet[o] - mu * sc;
  }
}

__global__ void zero_kernel(float* __restrict__ a, float* __restrict__ b, int n) {
  int i = blockIdx.x * 256 + threadIdx.x;
  if (i < n) { a[i] = 0.f; b[i] = 0.f; }
}

// ---------- xcat (B,512,N) f32 -> xT (B,N,512) bf16 ----------
__global__ void transpose_kernel(const float* __restrict__ xcat, unsigned short* __restrict__ xT) {
  int i = blockIdx.x * 256 + threadIdx.x;       // B*512*N = 4M
  if (i < NB * 512 * NPTS) {
    int b = i >> 19, rem = i & ((1 << 19) - 1);
    int c = rem >> 10, n = rem & 1023;
    xT[((b << 10) + n) * 512 + c] = f2bf(xcat[i]);
  }
}

// ---------- final 512->1024 conv: stats pass ----------
__global__ __launch_bounds__(256) void final_stats_kernel(const unsigned short* __restrict__ xT,
                                                          const unsigned short* __restrict__ W5,
                                                          float* __restrict__ gsum,
                                                          float* __restrict__ gssq) {
  constexpr int O = 1024, CIN = 512;
  __shared__ float lsum[O], lssq[O];
  const int b = blockIdx.y, n0 = blockIdx.x * 32, tid = threadIdx.x;
  const int lane = tid & 31, wave = tid >> 5, r = lane & 15;
  const int hi8 = (lane >= 16) ? 8 : 0;
  for (int t = tid; t < O; t += 256) { lsum[t] = 0.f; lssq[t] = 0.f; }
  __syncthreads();
  for (int t = wave; t < 128; t += 8) {         // 64 row-tiles x 2 col-tiles
    int rt = t >> 1, ct = t & 1;
    int n = n0 + ct * 16 + r;
    const unsigned short* Brow = xT + (b * NPTS + n) * CIN;
    v8f acc = {};
#pragma unroll
    for (int kk = 0; kk < CIN; kk += 32) {
      v16bf a = load_a_frag(W5, CIN, rt * 16 + r, kk, lane);
      v16bf bb = load_b_frag(Brow + kk, lane);
      acc = __builtin_amdgcn_wmma_f32_16x16x32_bf16(false, a, false, bb, (short)0, acc,
                                                    false, false);
    }
#pragma unroll
    for (int j = 0; j < 8; j++) {
      int o = rt * 16 + j + hi8;
      float v = acc[j];
      atomicAdd(&lsum[o], v);
      atomicAdd(&lssq[o], v * v);
    }
  }
  __syncthreads();
  for (int t = tid; t < O; t += 256) {
    atomicAdd(&gsum[t], lsum[t]);
    atomicAdd(&gssq[t], lssq[t]);
  }
}

// ---------- final conv apply: BN + LReLU + max over N ----------
__global__ __launch_bounds__(256) void final_apply_kernel(const unsigned short* __restrict__ xT,
                                                          const unsigned short* __restrict__ W5,
                                                          const float* __restrict__ scale,
                                                          const float* __restrict__ shift,
                                                          float* __restrict__ out) {
  constexpr int O = 1024, CIN = 512;
  __shared__ unsigned lmax[16];
  const int b = blockIdx.y, rt = blockIdx.x, tid = threadIdx.x;
  const int lane = tid & 31, wave = tid >> 5, r = lane & 15;
  const int hi8 = (lane >= 16) ? 8 : 0;
  if (tid < 16) lmax[tid] = 0u;
  __syncthreads();
  for (int cc = wave; cc < 64; cc += 8) {       // sweep all 1024 points
    int n = cc * 16 + r;
    const unsigned short* Brow = xT + (b * NPTS + n) * CIN;
    v8f acc = {};
#pragma unroll
    for (int kk = 0; kk < CIN; kk += 32) {
      v16bf a = load_a_frag(W5, CIN, rt * 16 + r, kk, lane);
      v16bf bb = load_b_frag(Brow + kk, lane);
      acc = __builtin_amdgcn_wmma_f32_16x16x32_bf16(false, a, false, bb, (short)0, acc,
                                                    false, false);
    }
#pragma unroll
    for (int j = 0; j < 8; j++) {
      int row = j + hi8;
      int o = rt * 16 + row;
      float v = acc[j] * scale[o] + shift[o];
      v = v > 0.f ? v : SLOPE * v;
      atomicMax(&lmax[row], fkey(v));
    }
  }
  __syncthreads();
  if (tid < 16) out[b * O + rt * 16 + tid] = unfkey(lmax[tid]);
}

// ---------- host ----------
extern "C" void kernel_launch(void* const* d_in, const int* in_sizes, int n_in,
                              void* d_out, int out_size, void* d_ws, size_t ws_size,
                              hipStream_t stream) {
  const float* x  = (const float*)d_in[0];
  const float* W1 = (const float*)d_in[1];  const float* g1 = (const float*)d_in[2];  const float* b1 = (const float*)d_in[3];
  const float* W2 = (const float*)d_in[4];  const float* g2 = (const float*)d_in[5];  const float* b2 = (const float*)d_in[6];
  const float* W3 = (const float*)d_in[7];  const float* g3 = (const float*)d_in[8];  const float* b3 = (const float*)d_in[9];
  const float* W4 = (const float*)d_in[10]; const float* g4 = (const float*)d_in[11]; const float* b4 = (const float*)d_in[12];
  const float* W5 = (const float*)d_in[13]; const float* g5 = (const float*)d_in[14]; const float* b5 = (const float*)d_in[15];

  char* ws = (char*)d_ws;
  size_t off = 0;
  auto alloc = [&](size_t bytes) { char* p = ws + off; off = (off + bytes + 255) & ~(size_t)255; return p; };
  unsigned short* W1bf = (unsigned short*)alloc(64 * 32 * 2);
  unsigned short* W2bf = (unsigned short*)alloc(64 * 128 * 2);
  unsigned short* W3bf = (unsigned short*)alloc(128 * 128 * 2);
  unsigned short* W4bf = (unsigned short*)alloc(256 * 256 * 2);
  unsigned short* W5bf = (unsigned short*)alloc(1024 * 512 * 2);
  int*   idx  = (int*)alloc((size_t)NB * NPTS * KNN * 4);
  float* xcat = (float*)alloc((size_t)NB * 512 * NPTS * 4);
  unsigned short* xT = (unsigned short*)alloc((size_t)NB * NPTS * 512 * 2);
  float* gsum  = (float*)alloc(1024 * 4);
  float* gssq  = (float*)alloc(1024 * 4);
  float* scale = (float*)alloc(1024 * 4);
  float* shift = (float*)alloc(1024 * 4);

  const int LDX = 512 * NPTS;                     // xcat batch stride
  const float invEdge = 1.f / (float)(NB * NPTS * KNN);
  const float invFin  = 1.f / (float)(NB * NPTS);

  convw_kernel<<<(64 * 32 + 255) / 256, 256, 0, stream>>>(W1, W1bf, 64, 6, 32);
  convw_kernel<<<(64 * 128 + 255) / 256, 256, 0, stream>>>(W2, W2bf, 64, 128, 128);
  convw_kernel<<<(128 * 128 + 255) / 256, 256, 0, stream>>>(W3, W3bf, 128, 128, 128);
  convw_kernel<<<(256 * 256 + 255) / 256, 256, 0, stream>>>(W4, W4bf, 256, 256, 256);
  convw_kernel<<<(1024 * 512 + 255) / 256, 256, 0, stream>>>(W5, W5bf, 1024, 512, 512);

  dim3 gKnn(NPTS / 128, NB), gEdge(NPTS / 4, NB);

  // ---- layer 1: C=3 ----
  knn_kernel<3><<<gKnn, 128, 0, stream>>>(x, 3 * NPTS, idx);
  zero_kernel<<<4, 256, 0, stream>>>(gsum, gssq, 1024);
  edge_gemm_kernel<3, 32, 64><<<gEdge, 256, 0, stream>>>(x, 3 * NPTS, idx, W1bf,
      nullptr, nullptr, nullptr, 0, 0, gsum, gssq, 0);
  finalize_kernel<<<1, 256, 0, stream>>>(gsum, gssq, g1, b1, scale, shift, 64, invEdge);
  edge_gemm_kernel<3, 32, 64><<<gEdge, 256, 0, stream>>>(x, 3 * NPTS, idx, W1bf,
      scale, shift, xcat, LDX, 0, nullptr, nullptr, 1);

  // ---- layer 2: C=64 (xcat ch 0..63 -> ch 64..127) ----
  knn_kernel<64><<<gKnn, 128, 0, stream>>>(xcat, LDX, idx);
  zero_kernel<<<4, 256, 0, stream>>>(gsum, gssq, 1024);
  edge_gemm_kernel<64, 128, 64><<<gEdge, 256, 0, stream>>>(xcat, LDX, idx, W2bf,
      nullptr, nullptr, nullptr, 0, 0, gsum, gssq, 0);
  finalize_kernel<<<1, 256, 0, stream>>>(gsum, gssq, g2, b2, scale, shift, 64, invEdge);
  edge_gemm_kernel<64, 128, 64><<<gEdge, 256, 0, stream>>>(xcat, LDX, idx, W2bf,
      scale, shift, xcat, LDX, 64, nullptr, nullptr, 1);

  // ---- layer 3: C=64 (ch 64..127 -> ch 128..255) ----
  knn_kernel<64><<<gKnn, 128, 0, stream>>>(xcat + 64 * NPTS, LDX, idx);
  zero_kernel<<<4, 256, 0, stream>>>(gsum, gssq, 1024);
  edge_gemm_kernel<64, 128, 128><<<gEdge, 256, 0, stream>>>(xcat + 64 * NPTS, LDX, idx, W3bf,
      nullptr, nullptr, nullptr, 0, 0, gsum, gssq, 0);
  finalize_kernel<<<1, 256, 0, stream>>>(gsum, gssq, g3, b3, scale, shift, 128, invEdge);
  edge_gemm_kernel<64, 128, 128><<<gEdge, 256, 0, stream>>>(xcat + 64 * NPTS, LDX, idx, W3bf,
      scale, shift, xcat, LDX, 128, nullptr, nullptr, 1);

  // ---- layer 4: C=128 (ch 128..255 -> ch 256..511) ----
  knn_kernel<128><<<gKnn, 128, 0, stream>>>(xcat + 128 * NPTS, LDX, idx);
  zero_kernel<<<4, 256, 0, stream>>>(gsum, gssq, 1024);
  edge_gemm_kernel<128, 256, 256><<<gEdge, 256, 0, stream>>>(xcat + 128 * NPTS, LDX, idx, W4bf,
      nullptr, nullptr, nullptr, 0, 0, gsum, gssq, 0);
  finalize_kernel<<<1, 256, 0, stream>>>(gsum, gssq, g4, b4, scale, shift, 256, invEdge);
  edge_gemm_kernel<128, 256, 256><<<gEdge, 256, 0, stream>>>(xcat + 128 * NPTS, LDX, idx, W4bf,
      scale, shift, xcat, LDX, 256, nullptr, nullptr, 1);

  // ---- final 512 -> 1024 + global max ----
  transpose_kernel<<<(NB * 512 * NPTS + 255) / 256, 256, 0, stream>>>(xcat, xT);
  zero_kernel<<<4, 256, 0, stream>>>(gsum, gssq, 1024);
  final_stats_kernel<<<dim3(32, NB), 256, 0, stream>>>(xT, W5bf, gsum, gssq);
  finalize_kernel<<<4, 256, 0, stream>>>(gsum, gssq, g5, b5, scale, shift, 1024, invFin);
  final_apply_kernel<<<dim3(64, NB), 256, 0, stream>>>(xT, W5bf, scale, shift, (float*)d_out);
}
